// AutoTailoredSmallDetector_7988639171218
// MI455X (gfx1250) — matrix operand order repaired and checked by reading, not compiled, not run on hardware
//
#include <hip/hip_runtime.h>
#include <hip/hip_bf16.h>
#include <math.h>

// ---------------------------------------------------------------------------
// CDNA5 WMMA types
// ---------------------------------------------------------------------------
typedef __attribute__((ext_vector_type(16))) __bf16 v16bf;
typedef __attribute__((ext_vector_type(8)))  __bf16 v8bf;
typedef __attribute__((ext_vector_type(8)))  float  v8f;

#define NPRI 5376
#define TOPK 200

static __device__ inline v16bf cat8(v8bf lo, v8bf hi)
{
    return __builtin_shufflevector(lo, hi, 0, 1, 2, 3, 4, 5, 6, 7,
                                           8, 9, 10, 11, 12, 13, 14, 15);
}

// ---------------------------------------------------------------------------
// conv1 (Cin=3, K=27): masked-gather implicit GEMM from f32 NCHW input,
// writes channels-last bf16 output [b][oh][ow][co].  One 16x16 tile / wave.
// ---------------------------------------------------------------------------
__global__ __launch_bounds__(32) void conv1_kernel(
    const float* __restrict__ x, const float* __restrict__ w,
    const float* __restrict__ bias, __bf16* __restrict__ y)
{
    const int lane = threadIdx.x;
    const int half = lane >> 4;
    const int l15  = lane & 15;

    const int nt = blockIdx.x;            // 16-pixel tile (npix = 16384)
    const int mt = blockIdx.y;            // 16-channel tile (Cout = 64)
    const int b  = blockIdx.z;

    const int p  = nt * 16 + l15;
    const int oh = p >> 7, ow = p & 127;  // Wout = 128
    const int m  = mt * 16 + l15;

    const long in_img = (long)b * 3 * 256 * 256;

    v16bf af, bf;
    #pragma unroll
    for (int i = 0; i < 16; ++i) {
        int k = ((i >> 3) << 4) + (half << 3) + (i & 7);
        af[i] = (__bf16)((k < 27) ? w[m * 27 + k] : 0.0f);
    }
    #pragma unroll
    for (int i = 0; i < 16; ++i) {
        int k = ((i >> 3) << 4) + (half << 3) + (i & 7);
        float v = 0.0f;
        if (k < 27) {
            int ci = k / 9; int r = k - ci * 9; int kh = r / 3; int kw = r - kh * 3;
            int ih = oh * 2 + kh, iw = ow * 2 + kw;   // stride 2, pad 0
            if (ih < 256 && iw < 256)
                v = x[in_img + ((long)ci * 256 + ih) * 256 + iw];
        }
        bf[i] = (__bf16)v;
    }

    v8f acc = {};
    acc = __builtin_amdgcn_wmma_f32_16x16x32_bf16(
              false, af, false, bf, (short)0, acc, false, false);

    // channels-last store: 8 consecutive out-channels per lane
    v8bf o;
    #pragma unroll
    for (int v = 0; v < 8; ++v) {
        int mm = mt * 16 + (half << 3) + v;
        o[v] = (__bf16)fmaxf(acc[v] + bias[mm], 0.0f);
    }
    *(v8bf*)(y + ((long)b * 16384 + p) * 64 + mt * 16 + (half << 3)) = o;
}

// ---------------------------------------------------------------------------
// Fast implicit-GEMM conv for Cin % 32 == 0.
//   input : channels-last bf16 [b][h][w][ci]
//   weights: pre-packed bf16 [Cout][tap][ci]  (tap-major, ci contiguous)
//   output: channels-last bf16 (OUT_F32=false) or f32 (heads, OUT_F32=true)
// One wave computes a 32(M) x 32(N) tile: 4 accumulators, 4 WMMAs/K-step.
// All fragment loads are contiguous 16/32-byte vector loads -> b128s.
// ---------------------------------------------------------------------------
template<int KS, bool OUT_F32>
__global__ __launch_bounds__(32) void conv_fast_kernel(
    const __bf16* __restrict__ x, const __bf16* __restrict__ wpk,
    const float* __restrict__ bias, void* __restrict__ yv,
    int Cin, int Hin, int Win, int Cout, int npix, int wshift,
    int stride, int pad, int do_relu)
{
    const int lane = threadIdx.x;
    const int half = lane >> 4;
    const int l15  = lane & 15;

    const int nt = blockIdx.x;
    const int mt = blockIdx.y;
    const int b  = blockIdx.z;

    const int wmask = (1 << wshift) - 1;
    const int KK2   = KS * KS;

    const int p0 = nt * 32 + l15, p1 = p0 + 16;
    const int oh0 = p0 >> wshift, ow0 = p0 & wmask;
    const int oh1 = p1 >> wshift, ow1 = p1 & wmask;

    const int m0 = mt * 32 + l15, m1 = m0 + 16;
    const bool m0ok = (m0 < Cout), m1ok = (m1 < Cout);
    const __bf16* wp0 = wpk + (long)m0 * KK2 * Cin;
    const __bf16* wp1 = wpk + (long)m1 * KK2 * Cin;

    const long img = (long)b * Hin * Win;

    v8f acc00 = {}, acc01 = {}, acc10 = {}, acc11 = {};
    const v16bf zv = {};

    #pragma unroll
    for (int kh = 0; kh < KS; ++kh) {
        #pragma unroll
        for (int kw = 0; kw < KS; ++kw) {
            const int tap = kh * KS + kw;
            const int ih0 = oh0 * stride + kh - pad, iw0 = ow0 * stride + kw - pad;
            const int ih1 = oh1 * stride + kh - pad, iw1 = ow1 * stride + kw - pad;
            const bool v0 = (ih0 >= 0) & (ih0 < Hin) & (iw0 >= 0) & (iw0 < Win);
            const bool v1 = (ih1 >= 0) & (ih1 < Hin) & (iw1 >= 0) & (iw1 < Win);
            const __bf16* bp0 = x + (img + (long)ih0 * Win + iw0) * Cin + (half << 4);
            const __bf16* bp1 = x + (img + (long)ih1 * Win + iw1) * Cin + (half << 4);
            const __bf16* ap0 = wp0 + (long)tap * Cin + (half << 4);
            const __bf16* ap1 = wp1 + (long)tap * Cin + (half << 4);

            for (int c0 = 0; c0 < Cin; c0 += 32) {
                // A fragments: two contiguous 8-element (16B) loads each
                v16bf a0 = m0ok ? cat8(*(const v8bf*)(ap0 + c0),
                                       *(const v8bf*)(ap0 + c0 + 8)) : zv;
                v16bf a1 = m1ok ? cat8(*(const v8bf*)(ap1 + c0),
                                       *(const v8bf*)(ap1 + c0 + 8)) : zv;
                // B fragments: one contiguous 16-element (32B) load each
                v16bf b0 = v0 ? *(const v16bf*)(bp0 + c0) : zv;
                v16bf b1 = v1 ? *(const v16bf*)(bp1 + c0) : zv;

                acc00 = __builtin_amdgcn_wmma_f32_16x16x32_bf16(
                            false, a0, false, b0, (short)0, acc00, false, false);
                acc01 = __builtin_amdgcn_wmma_f32_16x16x32_bf16(
                            false, a0, false, b1, (short)0, acc01, false, false);
                acc10 = __builtin_amdgcn_wmma_f32_16x16x32_bf16(
                            false, a1, false, b0, (short)0, acc10, false, false);
                acc11 = __builtin_amdgcn_wmma_f32_16x16x32_bf16(
                            false, a1, false, b1, (short)0, acc11, false, false);
            }
        }
    }

    // ---- channels-last stores: rows of D are consecutive M for fixed N ----
    const int  mba = mt * 32 + (half << 3);   // acc0x row base
    const int  mbb = mba + 16;                // acc1x row base
    const long row0 = ((long)b * npix + p0) * Cout;
    const long row1 = ((long)b * npix + p1) * Cout;

    #define EMIT(ACC, ROW, MB)                                                  \
        if ((MB) < Cout) {                                                      \
            if constexpr (OUT_F32) {                                            \
                float* y = (float*)yv;                                          \
                v8f o;                                                          \
                _Pragma("unroll")                                               \
                for (int v = 0; v < 8; ++v) {                                   \
                    float f = (ACC)[v] + bias[(MB) + v];                        \
                    o[v] = do_relu ? fmaxf(f, 0.0f) : f;                        \
                }                                                               \
                *(v8f*)(y + (ROW) + (MB)) = o;                                  \
            } else {                                                            \
                __bf16* y = (__bf16*)yv;                                        \
                v8bf o;                                                         \
                _Pragma("unroll")                                               \
                for (int v = 0; v < 8; ++v) {                                   \
                    float f = (ACC)[v] + bias[(MB) + v];                        \
                    o[v] = (__bf16)(do_relu ? fmaxf(f, 0.0f) : f);              \
                }                                                               \
                *(v8bf*)(y + (ROW) + (MB)) = o;                                 \
            }                                                                   \
        }

    EMIT(acc00, row0, mba)
    EMIT(acc01, row1, mba)
    EMIT(acc10, row0, mbb)
    EMIT(acc11, row1, mbb)
    #undef EMIT
}

// ---------------------------------------------------------------------------
// Weight packing: f32 OIHW -> bf16 [Cout][tap][ci]
// ---------------------------------------------------------------------------
__global__ void pack_w_kernel(const float* __restrict__ w, __bf16* __restrict__ wpk,
                              int Cout, int Cin, int KK2)
{
    int t = blockIdx.x * blockDim.x + threadIdx.x;
    int total = Cout * Cin * KK2;
    if (t >= total) return;
    int m   = t / (Cin * KK2);
    int r   = t - m * (Cin * KK2);
    int tap = r / Cin;
    int ci  = r - tap * Cin;
    wpk[t] = (__bf16)w[((long)m * Cin + ci) * KK2 + tap];
}

// ---------------------------------------------------------------------------
// Decode: heads are f32 channels-last [b][pix][ch] -> contiguous gathers
// ---------------------------------------------------------------------------
__global__ void decode_kernel(
    const float* __restrict__ loc0, const float* __restrict__ conf0,
    const float* __restrict__ loc1, const float* __restrict__ conf1,
    const float* __restrict__ loc2, const float* __restrict__ conf2,
    const float* __restrict__ priors,
    float* __restrict__ boxes, float* __restrict__ scores, int Btot)
{
    int t = blockIdx.x * blockDim.x + threadIdx.x;
    if (t >= Btot * NPRI) return;
    int b  = t / NPRI;
    int pr = t - b * NPRI;

    const float* loc; const float* conf; int npx, base;
    if (pr < 4096)      { loc = loc0; conf = conf0; npx = 1024; base = 0;    }
    else if (pr < 5120) { loc = loc1; conf = conf1; npx = 256;  base = 4096; }
    else                { loc = loc2; conf = conf2; npx = 64;   base = 5120; }

    int rel = pr - base;
    int pix = rel >> 2;
    int a   = rel & 3;

    const float* lp = loc  + ((long)b * npx + pix) * 16 + a * 4;
    const float* cp = conf + ((long)b * npx + pix) * 8  + a * 2;
    float l0 = lp[0], l1 = lp[1], l2 = lp[2], l3 = lp[3];
    float sc = 1.0f / (1.0f + expf(cp[0] - cp[1]));   // softmax class-1 prob

    float pcx = priors[pr * 4 + 0], pcy = priors[pr * 4 + 1];
    float pw  = priors[pr * 4 + 2], ph  = priors[pr * 4 + 3];
    float cx = pcx + l0 * 0.1f * pw;
    float cy = pcy + l1 * 0.1f * ph;
    float ww = pw * expf(l2 * 0.2f);
    float hh = ph * expf(l3 * 0.2f);

    boxes[(long)t * 4 + 0] = cx - 0.5f * ww;
    boxes[(long)t * 4 + 1] = cy - 0.5f * hh;
    boxes[(long)t * 4 + 2] = cx + 0.5f * ww;
    boxes[(long)t * 4 + 3] = cy + 0.5f * hh;
    scores[t] = sc;
}

// ---------------------------------------------------------------------------
// Per-image top-200 by iterative argmax (LDS-resident scores, stable ties)
// ---------------------------------------------------------------------------
__global__ __launch_bounds__(256) void topk_kernel(
    const float* __restrict__ scores, const float* __restrict__ boxes,
    float* __restrict__ tk_s, float* __restrict__ tk_b)
{
    __shared__ float ssc[NPRI];
    __shared__ float rv[256];
    __shared__ int   ri[256];

    const int b   = blockIdx.x;
    const int tid = threadIdx.x;

    for (int i = tid; i < NPRI; i += 256) ssc[i] = scores[(long)b * NPRI + i];
    __syncthreads();

    for (int it = 0; it < TOPK; ++it) {
        float best = -1e30f; int bi = 1 << 30;
        for (int i = tid; i < NPRI; i += 256) {
            float v = ssc[i];
            if (v > best || (v == best && i < bi)) { best = v; bi = i; }
        }
        rv[tid] = best; ri[tid] = bi;
        __syncthreads();
        for (int s = 128; s > 0; s >>= 1) {
            if (tid < s) {
                if (rv[tid + s] > rv[tid] ||
                    (rv[tid + s] == rv[tid] && ri[tid + s] < ri[tid])) {
                    rv[tid] = rv[tid + s]; ri[tid] = ri[tid + s];
                }
            }
            __syncthreads();
        }
        if (tid == 0) {
            int sel = ri[0];
            tk_s[b * TOPK + it] = rv[0];
            #pragma unroll
            for (int c = 0; c < 4; ++c)
                tk_b[(b * TOPK + it) * 4 + c] = boxes[((long)b * NPRI + sel) * 4 + c];
            ssc[sel] = -1e30f;
        }
        __syncthreads();
    }
}

// ---------------------------------------------------------------------------
// Per-image sequential NMS + stable-partition write of detections.
// d_out must be pre-zeroed: suppressed rows are zero rows in the reference.
// ---------------------------------------------------------------------------
__global__ __launch_bounds__(256) void nms_kernel(
    const float* __restrict__ tk_s, const float* __restrict__ tk_b,
    float* __restrict__ out)
{
    __shared__ float x1s[TOPK], y1s[TOPK], x2s[TOPK], y2s[TOPK];
    __shared__ float scs[TOPK], areas[TOPK];
    __shared__ int   keep[TOPK], pos[TOPK];

    const int b   = blockIdx.x;
    const int tid = threadIdx.x;

    for (int i = tid; i < TOPK; i += 256) {
        float s  = tk_s[b * TOPK + i];
        float x1 = tk_b[(b * TOPK + i) * 4 + 0];
        float y1 = tk_b[(b * TOPK + i) * 4 + 1];
        float x2 = tk_b[(b * TOPK + i) * 4 + 2];
        float y2 = tk_b[(b * TOPK + i) * 4 + 3];
        scs[i] = s; x1s[i] = x1; y1s[i] = y1; x2s[i] = x2; y2s[i] = y2;
        areas[i] = (x2 - x1) * (y2 - y1);
        keep[i]  = (s > 0.01f) ? 1 : 0;
    }
    __syncthreads();

    for (int i = 0; i < TOPK; ++i) {
        if (keep[i]) {
            for (int j = i + 1 + tid; j < TOPK; j += 256) {
                float ix1 = fmaxf(x1s[i], x1s[j]);
                float iy1 = fmaxf(y1s[i], y1s[j]);
                float ix2 = fminf(x2s[i], x2s[j]);
                float iy2 = fminf(y2s[i], y2s[j]);
                float inter = fmaxf(ix2 - ix1, 0.0f) * fmaxf(iy2 - iy1, 0.0f);
                float iou = inter / fmaxf(areas[i] + areas[j] - inter, 1e-9f);
                if (iou > 0.45f) keep[j] = 0;
            }
        }
        __syncthreads();
    }

    if (tid == 0) {
        int cnt = 0;
        for (int i = 0; i < TOPK; ++i) if (keep[i]) pos[i] = cnt++;
    }
    __syncthreads();

    for (int i = tid; i < TOPK; i += 256) {
        if (keep[i]) {
            int r = pos[i];
            long base = ((long)(b * 2 + 1) * TOPK + r) * 5;
            out[base + 0] = scs[i];
            out[base + 1] = x1s[i];
            out[base + 2] = y1s[i];
            out[base + 3] = x2s[i];
            out[base + 4] = y2s[i];
        }
    }
}

__global__ void zero_kernel(float* p, int n)
{
    int t = blockIdx.x * blockDim.x + threadIdx.x;
    if (t < n) p[t] = 0.0f;
}

// ---------------------------------------------------------------------------
// Host-side launchers
// ---------------------------------------------------------------------------
static inline void launch_pack(hipStream_t s, const float* w, __bf16* wpk,
                               int Cout, int Cin, int KK2)
{
    int total = Cout * Cin * KK2;
    pack_w_kernel<<<(total + 255) / 256, 256, 0, s>>>(w, wpk, Cout, Cin, KK2);
}

static inline void launch_fast(hipStream_t s,
    const __bf16* x, const __bf16* wpk, const float* bias, void* y,
    int B, int Cin, int Hin, int Win, int Cout, int Hout, int Wout,
    int ks, int stride, int pad, int relu, bool out_f32)
{
    int npix   = Hout * Wout;
    int wshift = __builtin_ctz(Wout);
    dim3 grid(npix / 32, (Cout + 31) / 32, B);
    if (ks == 1) {
        if (out_f32)
            conv_fast_kernel<1, true><<<grid, 32, 0, s>>>(x, wpk, bias, y,
                Cin, Hin, Win, Cout, npix, wshift, stride, pad, relu);
        else
            conv_fast_kernel<1, false><<<grid, 32, 0, s>>>(x, wpk, bias, y,
                Cin, Hin, Win, Cout, npix, wshift, stride, pad, relu);
    } else {
        if (out_f32)
            conv_fast_kernel<3, true><<<grid, 32, 0, s>>>(x, wpk, bias, y,
                Cin, Hin, Win, Cout, npix, wshift, stride, pad, relu);
        else
            conv_fast_kernel<3, false><<<grid, 32, 0, s>>>(x, wpk, bias, y,
                Cin, Hin, Win, Cout, npix, wshift, stride, pad, relu);
    }
}

extern "C" void kernel_launch(void* const* d_in, const int* in_sizes, int n_in,
                              void* d_out, int out_size, void* d_ws, size_t ws_size,
                              hipStream_t stream)
{
    (void)n_in; (void)ws_size;

    // Inputs in setup_inputs() dict (insertion) order
    const float* x   = (const float*)d_in[0];
    const float* w1  = (const float*)d_in[1];  const float* b1  = (const float*)d_in[2];
    const float* w2  = (const float*)d_in[3];  const float* b2  = (const float*)d_in[4];
    const float* w3  = (const float*)d_in[5];  const float* b3  = (const float*)d_in[6];
    const float* e1w = (const float*)d_in[7];  const float* e1b = (const float*)d_in[8];
    const float* e2w = (const float*)d_in[9];  const float* e2b = (const float*)d_in[10];
    const float* e3w = (const float*)d_in[11]; const float* e3b = (const float*)d_in[12];
    const float* e4w = (const float*)d_in[13]; const float* e4b = (const float*)d_in[14];
    const float* l0w = (const float*)d_in[15]; const float* l0b = (const float*)d_in[16];
    const float* c0w = (const float*)d_in[17]; const float* c0b = (const float*)d_in[18];
    const float* l1w = (const float*)d_in[19]; const float* l1b = (const float*)d_in[20];
    const float* c1w = (const float*)d_in[21]; const float* c1b = (const float*)d_in[22];
    const float* l2w = (const float*)d_in[23]; const float* l2b = (const float*)d_in[24];
    const float* c2w = (const float*)d_in[25]; const float* c2b = (const float*)d_in[26];
    const float* priors = (const float*)d_in[27];

    float* out = (float*)d_out;
    const int B = in_sizes[0] / (3 * 256 * 256);   // 32

    // ---- workspace bump allocator (256B aligned regions) ----
    char* cur = (char*)d_ws;
    auto alloc = [&](size_t bytes) -> void* {
        void* p = cur;
        cur += (bytes + 255) & ~(size_t)255;
        return p;
    };
    // activations (channels-last bf16)
    __bf16* act1 = (__bf16*)alloc((size_t)B * 16384 * 64 * 2);
    __bf16* act2 = (__bf16*)alloc((size_t)B * 4096 * 128 * 2);
    __bf16* act3 = (__bf16*)alloc((size_t)B * 1024 * 256 * 2);
    __bf16* e1o  = (__bf16*)alloc((size_t)B * 1024 * 128 * 2);
    __bf16* e2o  = (__bf16*)alloc((size_t)B * 256 * 256 * 2);
    __bf16* e3o  = (__bf16*)alloc((size_t)B * 256 * 128 * 2);
    __bf16* e4o  = (__bf16*)alloc((size_t)B * 64 * 256 * 2);
    // packed weights (bf16, [Cout][tap][ci])
    __bf16* w2p = (__bf16*)alloc((size_t)128 * 9 * 64 * 2);
    __bf16* w3p = (__bf16*)alloc((size_t)256 * 9 * 128 * 2);
    __bf16* e1p = (__bf16*)alloc((size_t)128 * 1 * 256 * 2);
    __bf16* e2p = (__bf16*)alloc((size_t)256 * 9 * 128 * 2);
    __bf16* e3p = (__bf16*)alloc((size_t)128 * 1 * 256 * 2);
    __bf16* e4p = (__bf16*)alloc((size_t)256 * 9 * 128 * 2);
    __bf16* l0p = (__bf16*)alloc((size_t)16 * 9 * 256 * 2);
    __bf16* c0p = (__bf16*)alloc((size_t)8 * 9 * 256 * 2);
    __bf16* l1p = (__bf16*)alloc((size_t)16 * 9 * 256 * 2);
    __bf16* c1p = (__bf16*)alloc((size_t)8 * 9 * 256 * 2);
    __bf16* l2p = (__bf16*)alloc((size_t)16 * 9 * 256 * 2);
    __bf16* c2p = (__bf16*)alloc((size_t)8 * 9 * 256 * 2);
    // head outputs (f32 channels-last [b][pix][ch])
    float* loc0 = (float*)alloc((size_t)B * 1024 * 16 * 4);
    float* cnf0 = (float*)alloc((size_t)B * 1024 * 8 * 4);
    float* loc1 = (float*)alloc((size_t)B * 256 * 16 * 4);
    float* cnf1 = (float*)alloc((size_t)B * 256 * 8 * 4);
    float* loc2 = (float*)alloc((size_t)B * 64 * 16 * 4);
    float* cnf2 = (float*)alloc((size_t)B * 64 * 8 * 4);
    // decode / nms scratch
    float* boxs = (float*)alloc((size_t)B * NPRI * 4 * 4);
    float* scrs = (float*)alloc((size_t)B * NPRI * 4);
    float* tks  = (float*)alloc((size_t)B * TOPK * 4);
    float* tkb  = (float*)alloc((size_t)B * TOPK * 4 * 4);

    // ---- zero output (suppressed rows & channel 0 stay zero) ----
    zero_kernel<<<(out_size + 255) / 256, 256, 0, stream>>>(out, out_size);

    // ---- pack all fast-path weights (tiny, one-time per call) ----
    launch_pack(stream, w2,  w2p, 128, 64, 9);
    launch_pack(stream, w3,  w3p, 256, 128, 9);
    launch_pack(stream, e1w, e1p, 128, 256, 1);
    launch_pack(stream, e2w, e2p, 256, 128, 9);
    launch_pack(stream, e3w, e3p, 128, 256, 1);
    launch_pack(stream, e4w, e4p, 256, 128, 9);
    launch_pack(stream, l0w, l0p, 16, 256, 9);
    launch_pack(stream, c0w, c0p,  8, 256, 9);
    launch_pack(stream, l1w, l1p, 16, 256, 9);
    launch_pack(stream, c1w, c1p,  8, 256, 9);
    launch_pack(stream, l2w, l2p, 16, 256, 9);
    launch_pack(stream, c2w, c2p,  8, 256, 9);

    // ---- conv1: gather path, f32 NCHW -> bf16 channels-last ----
    {
        dim3 grid(16384 / 16, 64 / 16, B);
        conv1_kernel<<<grid, 32, 0, stream>>>(x, w1, b1, act1);
    }
    // ---- backbone + extras (fast path, bf16 CL -> bf16 CL) ----
    launch_fast(stream, act1, w2p, b2,  act2, B,  64, 128, 128, 128, 64, 64, 3, 2, 0, 1, false);
    launch_fast(stream, act2, w3p, b3,  act3, B, 128,  64,  64, 256, 32, 32, 3, 2, 0, 1, false);
    launch_fast(stream, act3, e1p, e1b, e1o,  B, 256,  32,  32, 128, 32, 32, 1, 1, 0, 1, false);
    launch_fast(stream, e1o,  e2p, e2b, e2o,  B, 128,  32,  32, 256, 16, 16, 3, 2, 0, 1, false);
    launch_fast(stream, e2o,  e3p, e3b, e3o,  B, 256,  16,  16, 128, 16, 16, 1, 1, 0, 1, false);
    launch_fast(stream, e3o,  e4p, e4b, e4o,  B, 128,  16,  16, 256,  8,  8, 3, 2, 0, 1, false);
    // ---- heads (fast path, bf16 CL -> f32 CL), stride-1 SAME (pad 1), no ReLU
    launch_fast(stream, act3, l0p, l0b, loc0, B, 256, 32, 32, 16, 32, 32, 3, 1, 1, 0, true);
    launch_fast(stream, act3, c0p, c0b, cnf0, B, 256, 32, 32,  8, 32, 32, 3, 1, 1, 0, true);
    launch_fast(stream, e2o,  l1p, l1b, loc1, B, 256, 16, 16, 16, 16, 16, 3, 1, 1, 0, true);
    launch_fast(stream, e2o,  c1p, c1b, cnf1, B, 256, 16, 16,  8, 16, 16, 3, 1, 1, 0, true);
    launch_fast(stream, e4o,  l2p, l2b, loc2, B, 256,  8,  8, 16,  8,  8, 3, 1, 1, 0, true);
    launch_fast(stream, e4o,  c2p, c2b, cnf2, B, 256,  8,  8,  8,  8,  8, 3, 1, 1, 0, true);

    // ---- decode + score ----
    int ndec = B * NPRI;
    decode_kernel<<<(ndec + 255) / 256, 256, 0, stream>>>(
        loc0, cnf0, loc1, cnf1, loc2, cnf2, priors, boxs, scrs, B);

    // ---- top-200 per image ----
    topk_kernel<<<B, 256, 0, stream>>>(scrs, boxs, tks, tkb);

    // ---- NMS + write detections into channel 1 ----
    nms_kernel<<<B, 256, 0, stream>>>(tks, tkb, out);
}
